// CausalOPWFAttention_23828478558639
// MI455X (gfx1250) — matrix operand order repaired and checked by reading, not compiled, not run on hardware
//
#include <hip/hip_runtime.h>

// CDNA5 / gfx1250: wave32, WMMA 16x16x32 bf16 -> f32 accum, async LDS DMA.
typedef __attribute__((ext_vector_type(16))) __bf16 v16bf;
typedef __attribute__((ext_vector_type(8)))  __bf16 v8bf;
typedef __attribute__((ext_vector_type(8)))  float  v8f;
typedef __attribute__((ext_vector_type(4)))  int    v4i;

typedef __attribute__((address_space(1))) v4i as1_v4i;  // global int4
typedef __attribute__((address_space(3))) v4i as3_v4i;  // LDS int4

#if __has_builtin(__builtin_amdgcn_global_load_async_to_lds_b128)
#define ASYNC_LDS 1
#else
#define ASYNC_LDS 0
#endif

#define B_  4
#define N_  2048
#define D_  1024
#define H_  16
#define HD_ 64
#define NS_ 11

__device__ __forceinline__ float wred(float x) {
#pragma unroll
  for (int m = 16; m > 0; m >>= 1) x += __shfl_xor(x, m, 32);
  return x;
}

#if ASYNC_LDS
__device__ __forceinline__ void async_cp16(const __bf16* g, __bf16* lds) {
  __builtin_amdgcn_global_load_async_to_lds_b128(
      (as1_v4i*)g, (as3_v4i*)lds, 0, 0);
}
__device__ __forceinline__ void wait_async0() {
#if __has_builtin(__builtin_amdgcn_s_wait_asynccnt)
  __builtin_amdgcn_s_wait_asynccnt(0);
#else
  asm volatile("s_wait_asynccnt 0x0" ::: "memory");
#endif
}
#endif

// ---------------- fp32 -> bf16 ----------------
__global__ __launch_bounds__(256)
void cvt_bf16(const float* __restrict__ s, __bf16* __restrict__ d, long n) {
  long i = ((long)blockIdx.x * 256 + threadIdx.x) * 4;
  if (i + 3 < n) {
    float4 f = *(const float4*)(s + i);
    d[i + 0] = (__bf16)f.x; d[i + 1] = (__bf16)f.y;
    d[i + 2] = (__bf16)f.z; d[i + 3] = (__bf16)f.w;
  } else {
    for (long j = i; j < n; ++j) d[j] = (__bf16)s[j];
  }
}

// ---------------- WMMA GEMM: C[M,Nout] = A[M,K] * W[Nout,K]^T + bias ----------------
// Block tile 128x128, 8 waves, wave tile 32Mx64N (8 WMMAs / 32-K panel / wave).
// Double-buffered LDS; async global->LDS DMA when the toolchain exposes it.
// EPI==0 : Cf[m,n] = acc + bias[n]                              (fp32 out)
// EPI==1 : Cb[m,n] = bf16( aux[m,n] * sigmoid(acc + bias[n]) )  (gate fusion)
template <int EPI>
__global__ __launch_bounds__(256)
void gemm_bf16_wmma(const __bf16* __restrict__ A, const __bf16* __restrict__ W,
                    const float* __restrict__ bias,
                    float* __restrict__ Cf, __bf16* __restrict__ Cb,
                    const float* __restrict__ aux,
                    int M, int Nout, int K) {
  __shared__ __align__(64) __bf16 sA[2][128][32];  // 2 x 8 KB
  __shared__ __align__(64) __bf16 sB[2][128][32];  // 2 x 8 KB

  const int tid  = threadIdx.x;
  const int lane = tid & 31;
  const int wid  = tid >> 5;        // 8 waves
  const int l    = lane & 15;
  const int hi   = lane >> 4;

  const int m0 = blockIdx.y * 128;
  const int n0 = blockIdx.x * 128;
  const int wm = (wid & 3) * 32;    // wave M sub-tile (2x 16)
  const int wn = (wid >> 2) * 64;   // wave N sub-tile (4x 16)

  // loader mapping: 512 16B chunks per panel; this thread owns rows r0 and r0+64.
  const int r0 = tid >> 2;          // 0..63
  const int c0 = (tid & 3) * 8;     // 0,8,16,24
  const __bf16* Ap0 = A + (size_t)(m0 + r0) * K + c0;
  const __bf16* Ap1 = A + (size_t)(m0 + r0 + 64) * K + c0;
  const __bf16* Wp0 = W + (size_t)(n0 + r0) * K + c0;
  const __bf16* Wp1 = W + (size_t)(n0 + r0 + 64) * K + c0;

  v8f acc[2][4];
  {
    v8f zero = {};
#pragma unroll
    for (int mi = 0; mi < 2; ++mi)
#pragma unroll
      for (int ni = 0; ni < 4; ++ni) acc[mi][ni] = zero;
  }

#if ASYNC_LDS
  auto fill_async = [&](int k0, int buf) {
    async_cp16(Ap0 + k0, &sA[buf][r0][c0]);
    async_cp16(Ap1 + k0, &sA[buf][r0 + 64][c0]);
    async_cp16(Wp0 + k0, &sB[buf][r0][c0]);
    async_cp16(Wp1 + k0, &sB[buf][r0 + 64][c0]);
  };
  fill_async(0, 0);
  wait_async0();
#else
  {
    v8bf a0 = *(const v8bf*)Ap0, a1 = *(const v8bf*)Ap1;
    v8bf b0 = *(const v8bf*)Wp0, b1 = *(const v8bf*)Wp1;
    *(v8bf*)&sA[0][r0][c0]      = a0;
    *(v8bf*)&sA[0][r0 + 64][c0] = a1;
    *(v8bf*)&sB[0][r0][c0]      = b0;
    *(v8bf*)&sB[0][r0 + 64][c0] = b1;
  }
#endif
  __syncthreads();

  int cur = 0;
  for (int k0 = 0; k0 < K; k0 += 32) {
    const int  nxt  = cur ^ 1;
    const bool more = (k0 + 32) < K;

#if ASYNC_LDS
    if (more) fill_async(k0 + 32, nxt);   // DMA overlaps the WMMAs below
#else
    v8bf a0 = {}, a1 = {}, b0 = {}, b1 = {};
    if (more) {
      a0 = *(const v8bf*)(Ap0 + k0 + 32);
      a1 = *(const v8bf*)(Ap1 + k0 + 32);
      b0 = *(const v8bf*)(Wp0 + k0 + 32);
      b1 = *(const v8bf*)(Wp1 + k0 + 32);
      if (k0 + 64 < K) {
        __builtin_prefetch(Ap0 + k0 + 64, 0, 3);
        __builtin_prefetch(Wp0 + k0 + 64, 0, 3);
      }
    }
#endif

    // ---- compute on buffer `cur` ----
    // A fragment (16x32): lane l<16 row M=l holds K {0..15}; l+16 holds {16..31}
    v16bf af[2];
#pragma unroll
    for (int mi = 0; mi < 2; ++mi) {
      const __bf16* ar = &sA[cur][wm + mi * 16 + l][0];
      v8bf alo = *(const v8bf*)(ar + hi * 16);
      v8bf ahi = *(const v8bf*)(ar + hi * 16 + 8);
      af[mi] = __builtin_shufflevector(alo, ahi,
          0, 1, 2, 3, 4, 5, 6, 7, 8, 9, 10, 11, 12, 13, 14, 15);
    }
    // B fragment (32x16): column N = lane&15, K-half by lane>>4 (contiguous W^T row).
    v16bf bfr[4];
#pragma unroll
    for (int ni = 0; ni < 4; ++ni)
      bfr[ni] = *(const v16bf*)&sB[cur][wn + ni * 16 + l][hi * 16];

#pragma unroll
    for (int ni = 0; ni < 4; ++ni)
#pragma unroll
      for (int mi = 0; mi < 2; ++mi)
        acc[mi][ni] = __builtin_amdgcn_wmma_f32_16x16x32_bf16(
            false, af[mi], false, bfr[ni], (short)0, acc[mi][ni], false, false);

#if ASYNC_LDS
    wait_async0();                        // next buffer resident before the barrier
#else
    if (more) {
      *(v8bf*)&sA[nxt][r0][c0]      = a0;
      *(v8bf*)&sA[nxt][r0 + 64][c0] = a1;
      *(v8bf*)&sB[nxt][r0][c0]      = b0;
      *(v8bf*)&sB[nxt][r0 + 64][c0] = b1;
    }
#endif
    __syncthreads();
    cur = nxt;
  }

  // C/D layout: VGPR r -> row M = r + hi*8, col N = lane&15
#pragma unroll
  for (int mi = 0; mi < 2; ++mi) {
#pragma unroll
    for (int ni = 0; ni < 4; ++ni) {
      const int cn = n0 + wn + ni * 16 + l;
      const float bs = bias[cn];
#pragma unroll
      for (int r = 0; r < 8; ++r) {
        const int row = m0 + wm + mi * 16 + hi * 8 + r;
        float v = acc[mi][ni][r] + bs;
        if (EPI == 0) {
          Cf[(size_t)row * Nout + cn] = v;
        } else {
          float g = 1.f / (1.f + __expf(-v));
          Cb[(size_t)row * Nout + cn] = (__bf16)(aux[(size_t)row * Nout + cn] * g);
        }
      }
    }
  }
}

// ---------------- wavelet-scale diagonal attention ----------------
// One wave per (b, h, n) row; lanes own pairs of the 64-wide head dim.
__global__ __launch_bounds__(256)
void opwf_attn(const float* __restrict__ qkv,        // [B*N, 3*D]
               const float* __restrict__ scale_gain, // [NS, H]
               const float* __restrict__ Wq,         // [NS, HD]
               const float* __restrict__ ibyp,       // [H]
               float* __restrict__ gathered) {       // [B*N, D]
  const float D4c[4] = {0.4829629131445341f, 0.8365163037378079f,
                        0.2241438680420134f, -0.1294095225512604f};
  const int lane = threadIdx.x & 31;
  const int wid  = threadIdx.x >> 5;
  const long row = (long)blockIdx.x * 8 + wid;  // (b*H + h)*N + n
  const int n  = (int)(row % N_);
  const int bh = (int)(row / N_);
  const int h  = bh % H_;
  const int b  = bh / H_;
  const int d0 = lane * 2;
  const int hc = h * HD_;
  const size_t mrow = (size_t)(b * N_ + n) * (3 * D_);

  const float q0 = qkv[mrow + hc + d0];
  const float q1 = qkv[mrow + hc + d0 + 1];

  // gains = softmax_s( q . W_qscale[s] + scale_gain[s,h] )
  float g[NS_];
  float mx = -1e30f;
#pragma unroll
  for (int s = 0; s < NS_; ++s) {
    float p = q0 * Wq[s * HD_ + d0] + q1 * Wq[s * HD_ + d0 + 1];
    float o = wred(p) + scale_gain[s * H_ + h];
    g[s] = o;
    mx = fmaxf(mx, o);
  }
  float sum = 0.f;
#pragma unroll
  for (int s = 0; s < NS_; ++s) { g[s] = __expf(g[s] - mx); sum += g[s]; }
  const float inv = 1.f / sum;

  // local (offset 0) k, v and qk
  const float k0 = qkv[mrow + D_ + hc + d0];
  const float k1 = qkv[mrow + D_ + hc + d0 + 1];
  const float v0 = qkv[mrow + 2 * D_ + hc + d0];
  const float v1 = qkv[mrow + 2 * D_ + hc + d0 + 1];
  const float qk0 = wred(q0 * k0 + q1 * k1);

  float o0 = 0.f, o1 = 0.f;
#pragma unroll
  for (int j = 0; j < NS_; ++j) {
    const float gj = g[j] * inv;
    const int dsc = 1 << j;
    // tau = 0 (offset 0): reuse local qk/v
    const float c0 = gj * D4c[0];
    o0 += c0 * qk0 * v0;
    o1 += c0 * qk0 * v1;
    for (int tau = 1; tau < 4; ++tau) {
      const int off = dsc * tau;
      if (off >= N_ || n < off) continue;  // wave-uniform branch
      const size_t m2 = (size_t)(b * N_ + (n - off)) * (3 * D_);
      const float ks0 = qkv[m2 + D_ + hc + d0];
      const float ks1 = qkv[m2 + D_ + hc + d0 + 1];
      const float qk = wred(q0 * ks0 + q1 * ks1);
      const float c = gj * D4c[tau] * qk;
      o0 += c * qkv[m2 + 2 * D_ + hc + d0];
      o1 += c * qkv[m2 + 2 * D_ + hc + d0 + 1];
    }
  }
  const float bp = log1pf(__expf(ibyp[h]));  // softplus(identity_bypass[h])
  o0 += bp * qk0 * v0;
  o1 += bp * qk0 * v1;

  gathered[(size_t)(b * N_ + n) * D_ + hc + d0]     = o0;
  gathered[(size_t)(b * N_ + n) * D_ + hc + d0 + 1] = o1;
}

// ---------------- launch ----------------
extern "C" void kernel_launch(void* const* d_in, const int* in_sizes, int n_in,
                              void* d_out, int out_size, void* d_ws, size_t ws_size,
                              hipStream_t stream) {
  (void)in_sizes; (void)n_in; (void)out_size; (void)ws_size;
  const float* x          = (const float*)d_in[0];
  const float* W_qkv      = (const float*)d_in[1];
  const float* b_qkv      = (const float*)d_in[2];
  const float* W_out      = (const float*)d_in[3];
  const float* b_out      = (const float*)d_in[4];
  const float* W_gate     = (const float*)d_in[5];
  const float* b_gate     = (const float*)d_in[6];
  const float* scale_gain = (const float*)d_in[7];
  const float* W_qscale   = (const float*)d_in[8];
  const float* ident_byp  = (const float*)d_in[9];

  char* ws = (char*)d_ws;
  size_t off = 0;
  auto carve = [&](size_t bytes) -> void* {
    void* p = ws + off;
    off += (bytes + 255) & ~(size_t)255;
    return p;
  };

  const size_t MN = (size_t)B_ * N_;  // 8192 rows
  __bf16* xb     = (__bf16*)carve(MN * D_ * 2);               // 16 MB
  __bf16* wqkvb  = (__bf16*)carve((size_t)3 * D_ * D_ * 2);   //  6 MB
  __bf16* wgateb = (__bf16*)carve((size_t)D_ * D_ * 2);       //  2 MB
  __bf16* woutb  = (__bf16*)carve((size_t)D_ * D_ * 2);       //  2 MB
  float*  qkv    = (float*) carve(MN * 3 * D_ * 4);           // 96 MB
  float*  gath   = (float*) carve(MN * D_ * 4);               // 32 MB
  __bf16* gated  = (__bf16*)carve(MN * D_ * 2);               // 16 MB

  // 1) fp32 -> bf16 staging
  cvt_bf16<<<(int)((MN * D_) / 1024), 256, 0, stream>>>(x, xb, (long)(MN * D_));
  cvt_bf16<<<(3 * D_ * D_) / 1024, 256, 0, stream>>>(W_qkv, wqkvb, (long)(3 * D_ * D_));
  cvt_bf16<<<(D_ * D_) / 1024, 256, 0, stream>>>(W_gate, wgateb, (long)(D_ * D_));
  cvt_bf16<<<(D_ * D_) / 1024, 256, 0, stream>>>(W_out, woutb, (long)(D_ * D_));

  // 2) qkv = x @ W_qkv^T + b_qkv   (fp32 out for the wavelet stage)
  gemm_bf16_wmma<0><<<dim3((3 * D_) / 128, (int)(MN / 128)), 256, 0, stream>>>(
      xb, wqkvb, b_qkv, qkv, nullptr, nullptr, (int)MN, 3 * D_, D_);

  // 3) wavelet diagonal attention -> gathered
  opwf_attn<<<(B_ * H_ * N_) / 8, 256, 0, stream>>>(qkv, scale_gain, W_qscale,
                                                    ident_byp, gath);

  // 4) gated = bf16( gathered * sigmoid(x @ W_gate^T + b_gate) )   (fused epilogue)
  gemm_bf16_wmma<1><<<dim3(D_ / 128, (int)(MN / 128)), 256, 0, stream>>>(
      xb, wgateb, b_gate, nullptr, gated, gath, (int)MN, D_, D_);

  // 5) out = gated @ W_out^T + b_out
  gemm_bf16_wmma<0><<<dim3(D_ / 128, (int)(MN / 128)), 256, 0, stream>>>(
      gated, woutb, b_out, (float*)d_out, nullptr, nullptr, (int)MN, D_, D_);
}
